// GCNConvNet_81939386073493
// MI455X (gfx1250) — compile-verified, matrix operand order checked by reading, not run on hardware
//
#include <hip/hip_runtime.h>
#include <hip/hip_bf16.h>

typedef __attribute__((ext_vector_type(16))) __bf16 v16bf;
typedef __attribute__((ext_vector_type(8)))  __bf16 v8bf;
typedef __attribute__((ext_vector_type(2)))  __bf16 v2bf;
typedef __attribute__((ext_vector_type(8)))  float  v8f;

#define BM 64
#define BN 128
#define BK 32
#define LDA 40    // row stride in bf16 elems: 80B rows -> 16B-aligned, conflict-free b128
#define LDBT 40

// ---------------- graph preprocessing ----------------

__global__ void deg_init_kernel(float* __restrict__ deg, int n) {
    int i = blockIdx.x * blockDim.x + threadIdx.x;
    if (i < n) deg[i] = 1.0f;   // self loop contributes 1 to in-degree
}

__global__ void deg_accum_kernel(const int* __restrict__ col, float* __restrict__ deg, int e) {
    int i = blockIdx.x * blockDim.x + threadIdx.x;
    if (i < e) atomicAdd(&deg[col[i]], 1.0f);
}

__global__ void dis_kernel(const float* __restrict__ deg, float* __restrict__ dis, int n) {
    int i = blockIdx.x * blockDim.x + threadIdx.x;
    if (i < n) {
        float d = deg[i];
        dis[i] = (d > 0.0f) ? rsqrtf(fmaxf(d, 1e-12f)) : 0.0f;
    }
}

__global__ void norm_kernel(const int* __restrict__ row, const int* __restrict__ col,
                            const float* __restrict__ dis, float* __restrict__ nrm, int e) {
    int i = blockIdx.x * blockDim.x + threadIdx.x;
    if (i < e) nrm[i] = dis[row[i]] * dis[col[i]];
}

// ---------------- WMMA GEMM: H = bf16(reluA ? relu(X) : X) @ bf16(W) ----------------
// 256 threads = 8 waves (wave32). Block tile 64x128; wave tile 16x64 (4 WMMA tiles).
// Double-buffered LDS; fragments via conflict-free ds_load_b128; vectorized staging.
//
// Thread->element staging map (same for fast & tail paths):
//   A: thread t owns row r=t>>2, k-chunk c8=(t&3)*8 .. +7        (8 contiguous k)
//   B: group q in {t, t+256}: n=(q&31)*4 .. +3, k-pair kp=q>>5   (2 k-rows x 4 n)

__global__ __launch_bounds__(256) void gemm_bf16_wmma(const float* __restrict__ X,
                                                      const float* __restrict__ W,
                                                      float* __restrict__ H,
                                                      int M, int K, int Nout, int reluA) {
    __shared__ __align__(16) __bf16 As[2][BM * LDA];
    __shared__ __align__(16) __bf16 BsT[2][BN * LDBT];

    const int tid  = threadIdx.x;
    const int wave = tid >> 5;
    const int lane = tid & 31;
    const int wm   = wave & 3;
    const int wn   = wave >> 2;
    const int half = lane >> 4;
    const int l16  = lane & 15;

    const int m0 = blockIdx.x * BM;
    const int n0 = blockIdx.y * BN;

    // A staging coords
    const int ar  = tid >> 2;
    const int ac8 = (tid & 3) * 8;
    int gmA = m0 + ar; gmA = gmA < M ? gmA : M - 1;   // fully-OOB rows only

    float aval[8], bval[16];

    auto loadTile = [&](int k0) {
        const bool full = (k0 + BK) <= K;             // uniform
        // ---- A ----
        const float* xrow = X + (size_t)gmA * K;
        if (full) {
            float4 u = *(const float4*)(xrow + k0 + ac8);
            float4 w = *(const float4*)(xrow + k0 + ac8 + 4);
            aval[0] = u.x; aval[1] = u.y; aval[2] = u.z; aval[3] = u.w;
            aval[4] = w.x; aval[5] = w.y; aval[6] = w.z; aval[7] = w.w;
        } else {
#pragma unroll
            for (int j = 0; j < 8; ++j) {
                int gk = k0 + ac8 + j;
                int gkc = gk < K ? gk : K - 1;
                float v = xrow[gkc];
                aval[j] = (gk < K) ? v : 0.0f;
            }
        }
        if (reluA) {
#pragma unroll
            for (int j = 0; j < 8; ++j) aval[j] = fmaxf(aval[j], 0.0f);
        }
        // ---- B ----
        const bool fullB = full && (Nout >= 4);       // uniform
#pragma unroll
        for (int g = 0; g < 2; ++g) {
            int q  = tid + g * 256;
            int n  = (q & 31) * 4;
            int k  = (q >> 5) * 2;
            if (fullB) {
                int gn = n0 + n; gn = gn < (Nout - 3) ? gn : Nout - 4;  // whole-group clamp
                const float* wr = W + (size_t)(k0 + k) * Nout + gn;
                float4 ua = *(const float4*)wr;
                float4 ub = *(const float4*)(wr + Nout);
                bval[g * 8 + 0] = ua.x; bval[g * 8 + 1] = ub.x;
                bval[g * 8 + 2] = ua.y; bval[g * 8 + 3] = ub.y;
                bval[g * 8 + 4] = ua.z; bval[g * 8 + 5] = ub.z;
                bval[g * 8 + 6] = ua.w; bval[g * 8 + 7] = ub.w;
            } else {
#pragma unroll
                for (int j = 0; j < 4; ++j) {
#pragma unroll
                    for (int rr = 0; rr < 2; ++rr) {
                        int gk = k0 + k + rr;
                        int gkc = gk < K ? gk : K - 1;
                        int gn = n0 + n + j; gn = gn < Nout ? gn : Nout - 1;
                        float v = W[(size_t)gkc * Nout + gn];
                        bval[g * 8 + j * 2 + rr] = (gk < K) ? v : 0.0f;
                    }
                }
            }
        }
    };

    auto storeTile = [&](int buf) {
        // A: 8 contiguous bf16 -> one ds_store_b128
        v8bf av;
#pragma unroll
        for (int j = 0; j < 8; ++j) av[j] = (__bf16)aval[j];
        *(v8bf*)(&As[buf][ar * LDA + ac8]) = av;
        // B: (k,k+1) pairs -> ds_store_b32 each (transposed layout BsT[n][k])
#pragma unroll
        for (int g = 0; g < 2; ++g) {
            int q = tid + g * 256;
            int n = (q & 31) * 4;
            int k = (q >> 5) * 2;
#pragma unroll
            for (int j = 0; j < 4; ++j) {
                v2bf p;
                p.x = (__bf16)bval[g * 8 + j * 2 + 0];
                p.y = (__bf16)bval[g * 8 + j * 2 + 1];
                *(v2bf*)(&BsT[buf][(n + j) * LDBT + k]) = p;
            }
        }
    };

    // One 16x32 fragment = two contiguous 16B chunks per lane (ISA 7.12.2).
    auto loadFrag = [&](const __bf16* base, int frow) -> v16bf {
        const v8bf lo = *(const v8bf*)(base + frow + half * 8);
        const v8bf hi = *(const v8bf*)(base + frow + 16 + half * 8);
        return __builtin_shufflevector(lo, hi, 0, 1, 2, 3, 4, 5, 6, 7,
                                       8, 9, 10, 11, 12, 13, 14, 15);
    };

    v8f acc[4] = {{}, {}, {}, {}};

    loadTile(0);
    storeTile(0);
    __syncthreads();

    int cur = 0;
    for (int k0 = 0; k0 < K; k0 += BK) {
        const bool more = (k0 + BK) < K;
        if (more) loadTile(k0 + BK);   // global loads in flight during WMMAs below

        v16bf af  = loadFrag(&As[cur][0], (wm * 16 + l16) * LDA);
        v16bf bf0 = loadFrag(&BsT[cur][0], (wn * 64 +  0 + l16) * LDBT);
        v16bf bf1 = loadFrag(&BsT[cur][0], (wn * 64 + 16 + l16) * LDBT);
        v16bf bf2 = loadFrag(&BsT[cur][0], (wn * 64 + 32 + l16) * LDBT);
        v16bf bf3 = loadFrag(&BsT[cur][0], (wn * 64 + 48 + l16) * LDBT);

        acc[0] = __builtin_amdgcn_wmma_f32_16x16x32_bf16(false, af, false, bf0,
                                                         (short)0, acc[0], false, false);
        acc[1] = __builtin_amdgcn_wmma_f32_16x16x32_bf16(false, af, false, bf1,
                                                         (short)0, acc[1], false, false);
        acc[2] = __builtin_amdgcn_wmma_f32_16x16x32_bf16(false, af, false, bf2,
                                                         (short)0, acc[2], false, false);
        acc[3] = __builtin_amdgcn_wmma_f32_16x16x32_bf16(false, af, false, bf3,
                                                         (short)0, acc[3], false, false);

        if (more) {
            storeTile(cur ^ 1);
            __syncthreads();      // one barrier per k-step
            cur ^= 1;
        }
    }

#pragma unroll
    for (int s = 0; s < 4; ++s) {
        int gn = n0 + wn * 64 + s * 16 + l16;
#pragma unroll
        for (int r = 0; r < 8; ++r) {
            int gm = m0 + wm * 16 + r + 8 * half;
            if (gm < M && gn < Nout) H[(size_t)gm * Nout + gn] = acc[s][r];
        }
    }
}

// ---------------- aggregation ----------------

__global__ void self_bias_kernel(const float* __restrict__ h, const float* __restrict__ b,
                                 const float* __restrict__ dis, float* __restrict__ out,
                                 size_t total, int fshift) {
    size_t idx = (size_t)blockIdx.x * blockDim.x + threadIdx.x;
    if (idx < total) {
        int i = (int)(idx >> fshift);
        int c = (int)(idx & ((1u << fshift) - 1));
        float d = dis[i];
        out[idx] = b[c] + d * d * h[idx];
    }
}

template <int V>
__global__ void edge_scatter_kernel(const float* __restrict__ h, const int* __restrict__ row,
                                    const int* __restrict__ col, const float* __restrict__ nrm,
                                    float* __restrict__ out, size_t total, int cshift, int f) {
    size_t idx = (size_t)blockIdx.x * blockDim.x + threadIdx.x;
    if (idx >= total) return;
    int ei = (int)(idx >> cshift);
    int c  = (int)(idx & ((1u << cshift) - 1)) * V;
    float nv = nrm[ei];
    const float* hp = h + (size_t)row[ei] * f + c;
    float* op = out + (size_t)col[ei] * f + c;
    if (V == 4) {
        float4 hv = *(const float4*)hp;   // 16B-aligned: c%4==0, f%4==0
        atomicAdd(&op[0], nv * hv.x);
        atomicAdd(&op[1], nv * hv.y);
        atomicAdd(&op[2], nv * hv.z);
        atomicAdd(&op[3], nv * hv.w);
    } else {
#pragma unroll
        for (int j = 0; j < V; ++j) atomicAdd(&op[j], nv * hp[j]);
    }
}

// ---------------- host launch ----------------

static inline size_t align256(size_t x) { return (x + 255) & ~(size_t)255; }

extern "C" void kernel_launch(void* const* d_in, const int* in_sizes, int n_in,
                              void* d_out, int out_size, void* d_ws, size_t ws_size,
                              hipStream_t stream) {
    static const int DIMS[9] = {3, 64, 64, 64, 128, 1024, 512, 256, 2};

    const int N = in_sizes[0] / 3;
    const int E = in_sizes[1] / 2;

    const float* x0  = (const float*)d_in[0];
    const int*   ei  = (const int*)d_in[1];
    const int*   row = ei;       // edge_index[0]  (source)
    const int*   col = ei + E;   // edge_index[1]  (target)

    char* ws = (char*)d_ws;
    size_t off = 0;
    const size_t actBytes = align256((size_t)N * 1024 * sizeof(float));
    float* bufA = (float*)(ws + off); off += actBytes;
    float* bufB = (float*)(ws + off); off += actBytes;
    float* nrm  = (float*)(ws + off); off += align256((size_t)E * sizeof(float));
    float* dis  = (float*)(ws + off); off += align256((size_t)N * sizeof(float));
    float* deg  = (float*)(ws + off); off += align256((size_t)N * sizeof(float));
    (void)ws_size; (void)n_in; (void)out_size;

    const int TB = 256;

    deg_init_kernel<<<(N + TB - 1) / TB, TB, 0, stream>>>(deg, N);
    deg_accum_kernel<<<(E + TB - 1) / TB, TB, 0, stream>>>(col, deg, E);
    dis_kernel<<<(N + TB - 1) / TB, TB, 0, stream>>>(deg, dis, N);
    norm_kernel<<<(E + TB - 1) / TB, TB, 0, stream>>>(row, col, dis, nrm, E);

    const float* xin = x0;
    for (int layer = 0; layer < 8; ++layer) {
        const int Fin  = DIMS[layer];
        const int Fout = DIMS[layer + 1];
        const float* W = (const float*)d_in[2 + 2 * layer];
        const float* b = (const float*)d_in[3 + 2 * layer];

        dim3 grid((N + BM - 1) / BM, (Fout + BN - 1) / BN);
        gemm_bf16_wmma<<<grid, 256, 0, stream>>>(xin, W, bufB, N, Fin, Fout,
                                                 layer > 0 ? 1 : 0);

        float* out = (layer == 7) ? (float*)d_out : bufA;

        const int fshift = __builtin_ctz(Fout);
        size_t nf = (size_t)N << fshift;
        self_bias_kernel<<<(unsigned)((nf + TB - 1) / TB), TB, 0, stream>>>(bufB, b, dis, out, nf, fshift);

        const int V = (Fout >= 4) ? 4 : 2;
        const int cshift = fshift - ((V == 4) ? 2 : 1);
        size_t total = (size_t)E << cshift;
        if (V == 4)
            edge_scatter_kernel<4><<<(unsigned)((total + TB - 1) / TB), TB, 0, stream>>>(
                bufB, row, col, nrm, out, total, cshift, Fout);
        else
            edge_scatter_kernel<2><<<(unsigned)((total + TB - 1) / TB), TB, 0, stream>>>(
                bufB, row, col, nrm, out, total, cshift, Fout);

        xin = out;
    }
}
